// InfonceLoss_3100966387748
// MI455X (gfx1250) — matrix operand order repaired and checked by reading, not compiled, not run on hardware
//
#include <hip/hip_runtime.h>

// InfoNCE loss: B=8, T=256, V=32, D=256.
// scores[i,j,v,q,k] = sum_d feat[i,q,v,d]*faug[j,k,v,d]; E=exp(scores)
// ap[i,k] = sum_{v,q} E[i,i,v,q,k]        (diagonal blocks -> column sums)
// an[i,q] = sum_{j!=i,v,k} E[i,j,v,q,k]   (off-diagonal blocks -> row sums)
// out = mean_t sum_i -log(ap[i,t]/an[i,t])

#define B_ 8
#define T_ 256
#define V_ 32
#define D_ 256

#define TQ 128              // q-rows per workgroup tile
#define TK 128              // k-cols per workgroup tile
#define PADE 8              // bf16 pad elements per LDS row (16B-aligned rows)
#define LROW (D_ + PADE)    // 264 elements -> 528B row stride (33*16B)
#define VPW 4               // views per workgroup
#define LDS_BYTES (2 * TQ * LROW * 2)   // 135168 B dynamic LDS (<=320KB/WGP)

typedef __attribute__((ext_vector_type(8)))  __bf16 v8bf;
typedef __attribute__((ext_vector_type(16))) __bf16 v16bf;
typedef __attribute__((ext_vector_type(8)))  float  v8f;

union Frag16 { v16bf f; v8bf h[2]; };
union BF4    { __bf16 b[4]; unsigned long long u; };

__global__ void infonce_init(float* __restrict__ p, int n) {
    int t = blockIdx.x * 256 + threadIdx.x;
    if (t < n) p[t] = 0.0f;
}

__global__ __launch_bounds__(256)
void infonce_main(const float* __restrict__ feat, const float* __restrict__ faug,
                  float* __restrict__ ap, float* __restrict__ an)
{
    extern __shared__ __bf16 smem[];
    __bf16* Qs = smem;                 // [TQ][LROW]
    __bf16* Ks = smem + TQ * LROW;     // [TK][LROW]

    int idx = blockIdx.x;
    const int vg = idx & 7;  idx >>= 3;   // v-group (8)
    const int kt = idx & 1;  idx >>= 1;   // k tile (2)
    const int qt = idx & 1;  idx >>= 1;   // q tile (2)
    const int j  = idx & 7;  idx >>= 3;   // key batch
    const int i  = idx;                   // query batch

    const int q0 = qt * TQ;
    const int k0 = kt * TK;

    const int wave  = threadIdx.x >> 5;   // 0..7 (wave32)
    const int lane  = threadIdx.x & 31;
    const int lrow  = lane & 15;
    const int lhalf = lane >> 4;

    // 4x2 wave grid: each wave owns a 32x64 sub-tile (2 A-frags x 4 B-frags).
    const int rb = wave & 3;              // row block: rows rb*32 .. +32
    const int cb = wave >> 2;             // col block: cols cb*64 .. +64

    const bool isDiag = (i == j);

    float rowAcc[2][8];
    float colAcc[4];
#pragma unroll
    for (int ai = 0; ai < 2; ++ai)
#pragma unroll
        for (int r = 0; r < 8; ++r) rowAcc[ai][r] = 0.0f;
#pragma unroll
    for (int bi = 0; bi < 4; ++bi) colAcc[bi] = 0.0f;

    const v8f zacc = {};

    for (int vv = 0; vv < VPW; ++vv) {
        const int v = vg * VPW + vv;

        __syncthreads();   // previous compute phase done before overwriting LDS
        // Stage full-depth f32 -> bf16 panels into LDS (coalesced float4 loads).
        {
            const int baseQ = ((i * T_ + q0) * V_ + v) * D_;
            const int baseK = ((j * T_ + k0) * V_ + v) * D_;
            const int rowStride = V_ * D_;
#pragma unroll 4
            for (int it = 0; it < (TQ * D_ / 4) / 256; ++it) {   // 32 iters
                const int u  = it * 256 + threadIdx.x;
                const int rr = u >> 6;                 // 64 float4 per row
                const int d  = (u & 63) << 2;
                float4 q4 = *(const float4*)(feat + baseQ + rr * rowStride + d);
                float4 k4 = *(const float4*)(faug + baseK + rr * rowStride + d);
                BF4 qb, kb;
                qb.b[0] = (__bf16)q4.x; qb.b[1] = (__bf16)q4.y;
                qb.b[2] = (__bf16)q4.z; qb.b[3] = (__bf16)q4.w;
                kb.b[0] = (__bf16)k4.x; kb.b[1] = (__bf16)k4.y;
                kb.b[2] = (__bf16)k4.z; kb.b[3] = (__bf16)k4.w;
                *(unsigned long long*)&Qs[rr * LROW + d] = qb.u;
                *(unsigned long long*)&Ks[rr * LROW + d] = kb.u;
            }
        }
        __syncthreads();

        v8f acc[2][4];
#pragma unroll
        for (int ai = 0; ai < 2; ++ai)
#pragma unroll
            for (int bi = 0; bi < 4; ++bi) acc[ai][bi] = zacc;

        // Per-lane fragment bases (ISA layouts):
        // A 16x32 bf16: lane holds row lrow, K chunks {kk+lhalf*8..+7},{kk+16+lhalf*8..+7}
        // B 32x16 bf16: lane holds col lrow (= key row), K chunks {kk+lhalf*8..+7},{kk+16+lhalf*8..+7}
        const __bf16* qbase = Qs + (rb * 32 + lrow) * LROW + lhalf * 8;
        const __bf16* kbase = Ks + (cb * 64 + lrow) * LROW + lhalf * 8;

#pragma unroll
        for (int kk = 0; kk < D_; kk += 32) {
            Frag16 a[2], b[4];
#pragma unroll
            for (int ai = 0; ai < 2; ++ai) {
                const __bf16* p = qbase + ai * 16 * LROW + kk;
                a[ai].h[0] = *(const v8bf*)(p);
                a[ai].h[1] = *(const v8bf*)(p + 16);
            }
#pragma unroll
            for (int bi = 0; bi < 4; ++bi) {
                const __bf16* p = kbase + bi * 16 * LROW + kk;
                b[bi].h[0] = *(const v8bf*)(p);
                b[bi].h[1] = *(const v8bf*)(p + 16);
            }
#pragma unroll
            for (int ai = 0; ai < 2; ++ai)
#pragma unroll
                for (int bi = 0; bi < 4; ++bi)
                    acc[ai][bi] = __builtin_amdgcn_wmma_f32_16x16x32_bf16(
                        false, a[ai].f, false, b[bi].f, (short)0, acc[ai][bi],
                        false, false);
        }

        // exp() once per score; fold into per-lane row/col partial sums.
#pragma unroll
        for (int ai = 0; ai < 2; ++ai)
#pragma unroll
            for (int bi = 0; bi < 4; ++bi) {
                float cs = 0.0f;
#pragma unroll
                for (int r = 0; r < 8; ++r) {
                    float e = __expf(acc[ai][bi][r]);
                    rowAcc[ai][r] += e;
                    cs += e;
                }
                colAcc[bi] += cs;
            }
    }

    // C layout: VGPR r, lanes 0-15 -> (M=r, N=lane); lanes 16-31 -> (M=8+r, N=lane-16).
    if (isDiag) {
        // column sums over q -> ap[i, k]: combine lane l with l^16; lanes 0-15 emit.
        float* apRow = ap + i * T_ + k0 + cb * 64;
#pragma unroll
        for (int bi = 0; bi < 4; ++bi) {
            float c = colAcc[bi] + __shfl_xor(colAcc[bi], 16, 32);
            if (lhalf == 0)
                atomicAdd(&apRow[bi * 16 + lrow], c);
        }
    } else {
        // row sums over k -> an[i, q]: butterfly within each 16-lane half.
        float* anRow = an + i * T_ + q0 + rb * 32;
#pragma unroll
        for (int ai = 0; ai < 2; ++ai)
#pragma unroll
            for (int r = 0; r < 8; ++r) {
                float s = rowAcc[ai][r];
                s += __shfl_xor(s, 1, 32);
                s += __shfl_xor(s, 2, 32);
                s += __shfl_xor(s, 4, 32);
                s += __shfl_xor(s, 8, 32);
                if (lrow == 0)
                    atomicAdd(&anRow[ai * 16 + lhalf * 8 + r], s);
            }
    }
}

__global__ __launch_bounds__(256)
void infonce_finalize(const float* __restrict__ ap, const float* __restrict__ an,
                      float* __restrict__ out)
{
    __shared__ float red[256];
    const int t = threadIdx.x;
    float s = 0.0f;
#pragma unroll
    for (int i = 0; i < B_; ++i)
        s += -(__logf(ap[i * T_ + t]) - __logf(an[i * T_ + t]));
    red[t] = s;
    __syncthreads();
    for (int off = 128; off > 0; off >>= 1) {
        if (t < off) red[t] += red[t + off];
        __syncthreads();
    }
    if (t == 0) out[0] = red[0] / (float)T_;
}

extern "C" void kernel_launch(void* const* d_in, const int* in_sizes, int n_in,
                              void* d_out, int out_size, void* d_ws, size_t ws_size,
                              hipStream_t stream)
{
    (void)in_sizes; (void)n_in; (void)out_size; (void)ws_size;
    const float* feat = (const float*)d_in[0];
    const float* faug = (const float*)d_in[1];
    float* ap  = (float*)d_ws;            // [B_, T_]
    float* an  = ap + B_ * T_;            // [B_, T_]
    float* out = (float*)d_out;

    // Allow >64KB dynamic LDS (gfx1250 WGP has 320KB). Host-side attr set is
    // not a stream op -> graph-capture safe; harmless if it no-ops.
    static_assert(LDS_BYTES <= 320 * 1024, "LDS budget");
    (void)hipFuncSetAttribute((const void*)infonce_main,
                              hipFuncAttributeMaxDynamicSharedMemorySize,
                              LDS_BYTES);

    infonce_init<<<(2 * B_ * T_ + 255) / 256, 256, 0, stream>>>(ap, 2 * B_ * T_);

    const int nWG = B_ * B_ * (T_ / TQ) * (T_ / TK) * (V_ / VPW);  // 2048
    infonce_main<<<nWG, 256, LDS_BYTES, stream>>>(feat, faug, ap, an);

    infonce_finalize<<<1, 256, 0, stream>>>(ap, an, out);
}
